// AttentionSimilarity_56255481643431
// MI455X (gfx1250) — matrix operand order repaired and checked by reading, not compiled, not run on hardware
//
#include <hip/hip_runtime.h>
#include <hip/hip_bf16.h>

typedef __attribute__((ext_vector_type(16))) _Float16 v16h;
typedef __attribute__((ext_vector_type(8)))  _Float16 v8h;
typedef __attribute__((ext_vector_type(8)))  float    v8f;
typedef __attribute__((ext_vector_type(4)))  int      i32x4;

#define B_   64
#define C_   512
#define N_   100
#define IN_  64
#define M_   (B_ * N_)     // 6400 rows total
#define NPAD 112           // 100 padded to 7*16
#define KPAD 128           // 112 padded to 4*32 for P@V

// ---------------------------------------------------------------------------
// CDNA5 async global->LDS copy (GLOBAL_LOAD_ASYNC_TO_LDS_B128, ASYNCcnt).
// ---------------------------------------------------------------------------
#if __has_builtin(__builtin_amdgcn_global_load_async_to_lds_b128)
#define HAVE_ASYNC_LDS 1
#endif

typedef __attribute__((address_space(1))) i32x4* gv4_t;
typedef __attribute__((address_space(3))) i32x4* lv4_t;

__device__ inline void g2l_b128(const _Float16* __restrict__ g,
                                _Float16* __restrict__ l) {
#ifdef HAVE_ASYNC_LDS
  gv4_t gp = (gv4_t)(unsigned long long)g;            // generic == global addr
  lv4_t lp = (lv4_t)(unsigned)(unsigned long long)l;  // low 32 bits = LDS offs
  __builtin_amdgcn_global_load_async_to_lds_b128(gp, lp, 0, 0);
#else
  *(uint4*)l = *(const uint4*)g;
#endif
}

__device__ inline void wait_async() {
#if __has_builtin(__builtin_amdgcn_s_wait_asynccnt)
  __builtin_amdgcn_s_wait_asynccnt(0);
#elif defined(HAVE_ASYNC_LDS)
  asm volatile("s_wait_asynccnt 0x0" ::: "memory");
#endif
}

// ---------------------------------------------------------------------------
// WMMA fragment helpers (wave32, v_wmma_f32_16x16x32_f16).
// Per the ISA VGPR layouts, both loaders below are fully contiguous per lane:
//   A    : two 16-byte runs  (K = kh+0..7 and K = 16+kh..16+kh+7, kh in {0,8})
//   B^T  : two 16-byte runs  (K = kh+0..7 and K = 16+kh..16+kh+7, kh in {0,8})
// so each fragment is exactly 2 x ds_load_b128.
// ---------------------------------------------------------------------------
__device__ inline v16h frag_A(const _Float16* src, int ld) {
  const int l = threadIdx.x & 31;
  const _Float16* base = src + (l & 15) * ld + ((l >> 4) ? 8 : 0);
  const v8h lo = *(const v8h*)base;
  const v8h hi = *(const v8h*)(base + 16);
  return __builtin_shufflevector(lo, hi, 0, 1, 2, 3, 4, 5, 6, 7, 8, 9, 10, 11,
                                 12, 13, 14, 15);
}

// B fragment for X * Y^T where Y is row-major [n][K] f16 (lane = column n)
__device__ inline v16h frag_B_T(const _Float16* Y, int ld) {
  const int l = threadIdx.x & 31;
  const _Float16* base = Y + (l & 15) * ld + ((l >> 4) ? 8 : 0);
  const v8h lo = *(const v8h*)base;
  const v8h hi = *(const v8h*)(base + 16);
  return __builtin_shufflevector(lo, hi, 0, 1, 2, 3, 4, 5, 6, 7, 8, 9, 10, 11,
                                 12, 13, 14, 15);
}

// C/D layout: lane l holds col = l&15, VGPR r holds row = r + ((l>=16)?8:0)
__device__ inline void store_D_f32(float* dst, int ld, v8f c) {
  const int l   = threadIdx.x & 31;
  const int col = l & 15;
  const int rhi = (l >> 4) ? 8 : 0;
#pragma unroll
  for (int r = 0; r < 8; ++r) dst[(r + rhi) * ld + col] = c[r];
}

__device__ inline v8f wmma32(v16h a, v16h b, v8f c) {
  return __builtin_amdgcn_wmma_f32_16x16x32_f16(false, a, false, b, (short)0, c,
                                                false, false);
}

// ---------------------------------------------------------------------------
// Kernel 1: pack [B,C,HW] fp32 -> row-major [M_, C] f16 (both feature sets)
// ---------------------------------------------------------------------------
__global__ __launch_bounds__(256) void pack_kernel(const float* __restrict__ fa,
                                                   const float* __restrict__ fb,
                                                   _Float16* __restrict__ Xa,
                                                   _Float16* __restrict__ Xb) {
  const size_t idx = (size_t)blockIdx.x * 256 + threadIdx.x;
  if (idx >= (size_t)M_ * C_) return;
  const int r = (int)(idx >> 9);
  const int c = (int)(idx & 511);
  const int b = r / N_;
  const int n = r % N_;
  const size_t src = ((size_t)b * C_ + c) * N_ + n;
  Xa[idx] = (_Float16)fa[src];
  Xb[idx] = (_Float16)fb[src];
}

// ---------------------------------------------------------------------------
// Kernel 2: convert fp32 weights -> f16 TRANSPOSED (one-time, L2-resident).
// W1t = [Wq1^T|Wk1^T|Wv1^T]  (3 x [512 n][512 k])
// W2t = [Wq2^T|Wk2^T|Wv2^T]  (3 x [ 64 n][512 k])
// Transposed layout makes every projector B-fragment a contiguous B^T load.
// ---------------------------------------------------------------------------
__global__ __launch_bounds__(256) void cvt_w_kernel(
    const float* __restrict__ Wq1, const float* __restrict__ Wk1,
    const float* __restrict__ Wv1, const float* __restrict__ Wq2,
    const float* __restrict__ Wk2, const float* __restrict__ Wv2,
    _Float16* __restrict__ W1t, _Float16* __restrict__ W2t) {
  const int idx = blockIdx.x * 256 + threadIdx.x;
  if (idx < 3 * C_ * C_) {
    const int h = idx / (C_ * C_), r = idx % (C_ * C_);
    const int k = r / C_, n = r % C_;  // source W1[k][n]
    const float* s = (h == 0) ? Wq1 : (h == 1) ? Wk1 : Wv1;
    W1t[(size_t)h * C_ * C_ + (size_t)n * C_ + k] = (_Float16)s[r];
  }
  if (idx < 3 * C_ * IN_) {
    const int h = idx / (C_ * IN_), r = idx % (C_ * IN_);
    const int k = r / IN_, n = r % IN_;  // source W2[k][n]
    const float* s = (h == 0) ? Wq2 : (h == 1) ? Wk2 : Wv2;
    W2t[(size_t)h * IN_ * C_ + (size_t)n * C_ + k] = (_Float16)s[r];
  }
}

// ---------------------------------------------------------------------------
// Kernel 3: fused projector  out = f16( ReLU(X @ W1) @ W2 )
// grid = (400 row-tiles, 6 combos); 256 threads = 8 waves.
// Wave w keeps 4 persistent accumulators (cols 64w..64w+63); W1^T streamed
// through LDS in [512 n][32 k] slabs via async-to-LDS (16B chunks).
// ---------------------------------------------------------------------------
__global__ __launch_bounds__(256) void proj_kernel(
    const _Float16* __restrict__ Xa, const _Float16* __restrict__ Xb,
    const _Float16* __restrict__ W1t, const _Float16* __restrict__ W2t,
    _Float16* __restrict__ out6) {
  __shared__ _Float16 bufA[16 * C_];  // phase1: Xs    | phase2: W2 slab [64][32]
  __shared__ _Float16 bufB[32 * C_];  // phase1: W1 slab [512][32] | phase2: Hs
  _Float16* Xs  = bufA;
  _Float16* Ws  = bufB;
  _Float16* Hs  = bufB;
  _Float16* W2s = bufA;

  const int rt    = blockIdx.x;
  const int combo = blockIdx.y;
  const int srcB  = combo / 3;
  const int head  = combo % 3;
  const _Float16* X  = srcB ? Xb : Xa;
  const _Float16* W1 = W1t + (size_t)head * C_ * C_;
  const _Float16* W2 = W2t + (size_t)head * IN_ * C_;
  _Float16* out = out6 + (size_t)combo * M_ * IN_;

  const int tid  = threadIdx.x;
  const int wave = tid >> 5;
  const int l    = tid & 31;
  const int col  = l & 15;
  const int rhi  = (l >> 4) ? 8 : 0;

  // stage X tile (contiguous 16 KB) via async-to-LDS
  for (int i = tid; i < (16 * C_) / 8; i += 256)
    g2l_b128(X + (size_t)rt * 16 * C_ + i * 8, Xs + i * 8);
  wait_async();
  __syncthreads();

  // ---- GEMM1: hidden = ReLU(X @ W1), 4 column tiles per wave -------------
  v8f acc[4] = {{}, {}, {}, {}};
  for (int k0 = 0; k0 < C_; k0 += 32) {
    // slab: W1^T[n][k0..k0+31] for n = 0..511  (2048 x 16B async chunks)
    for (int i = tid; i < (32 * C_) / 8; i += 256) {
      const int n = i >> 2, c = i & 3;
      g2l_b128(W1 + (size_t)n * C_ + k0 + c * 8, Ws + n * 32 + c * 8);
    }
    wait_async();
    __syncthreads();
    const v16h a = frag_A(Xs + k0, C_);
#pragma unroll
    for (int j = 0; j < 4; ++j) {
      const v16h b = frag_B_T(Ws + (wave * 4 + j) * 16 * 32, 32);
      acc[j] = wmma32(a, b, acc[j]);
    }
    __syncthreads();  // slab reuse
  }
  // ReLU -> Hs (aliases Ws; all consumers are past the last barrier)
#pragma unroll
  for (int j = 0; j < 4; ++j) {
    const int cb = (wave * 4 + j) * 16;
#pragma unroll
    for (int r = 0; r < 8; ++r)
      Hs[(r + rhi) * C_ + cb + col] = (_Float16)fmaxf(acc[j][r], 0.0f);
  }
  __syncthreads();

  // ---- GEMM2: out = Hs @ W2, one 16x16 tile per wave (waves 0..3) --------
  v8f acc2 = {};
  for (int k0 = 0; k0 < C_; k0 += 32) {
    // slab: W2^T[n][k0..k0+31] for n = 0..63  (256 x 16B async chunks)
    for (int i = tid; i < (32 * IN_) / 8; i += 256) {
      const int n = i >> 2, c = i & 3;
      g2l_b128(W2 + (size_t)n * C_ + k0 + c * 8, W2s + n * 32 + c * 8);
    }
    wait_async();
    __syncthreads();
    if (wave < 4) {
      const v16h a = frag_A(Hs + k0, C_);
      const v16h b = frag_B_T(W2s + wave * 16 * 32, 32);
      acc2 = wmma32(a, b, acc2);
    }
    __syncthreads();
  }
  if (wave < 4) {
#pragma unroll
    for (int r = 0; r < 8; ++r)
      out[((size_t)rt * 16 + r + rhi) * IN_ + wave * 16 + col] =
          (_Float16)acc2[r];
  }
}

// ---------------------------------------------------------------------------
// Kernel 4: per (p,q) pair: attention (normalization-free softmax) + cosine.
// ~142 KB dynamic LDS (CDNA5: 320 KB/workgroup). V is transposed into LDS
// (Vt[i][m]) so P@V B-fragments are contiguous B^T loads too.
// ---------------------------------------------------------------------------
__global__ __launch_bounds__(256) void attn_kernel(
    const _Float16* __restrict__ qa, const _Float16* __restrict__ ka,
    const _Float16* __restrict__ va, const _Float16* __restrict__ qb,
    const _Float16* __restrict__ kb, const _Float16* __restrict__ vb,
    float* __restrict__ out) {
  extern __shared__ char smem[];
  _Float16* Qh = (_Float16*)smem;        // 112*64
  _Float16* Kh = Qh + NPAD * IN_;        // 112*64
  _Float16* Vt = Kh + NPAD * IN_;        // 64*128 (transposed V)
  _Float16* Vo = Vt + IN_ * KPAD;        // 112*64
  float*    S  = (float*)(Vo + NPAD * IN_);     // 112*128 f32
  _Float16* P  = (_Float16*)(S + NPAD * KPAD);  // 112*128 f16
  float*    AL = S;                              // aliased after P is built
  __shared__ float accRed;

  const int q    = blockIdx.x;
  const int p    = blockIdx.y;
  const int tid  = threadIdx.x;
  const int wave = tid >> 5;
  const float scale = 0.125f;  // 1/sqrt(64)

  if (tid == 0) accRed = 0.0f;

  for (int pass = 0; pass < 2; ++pass) {
    const _Float16 *Qg, *Kg, *Vg, *Og;
    if (pass == 0) {  // aligned_a: qb[p] vs ka[q]/va[q]; cosine vs vb[p]
      Qg = qb + (size_t)p * N_ * IN_;
      Kg = ka + (size_t)q * N_ * IN_;
      Vg = va + (size_t)q * N_ * IN_;
      Og = vb + (size_t)p * N_ * IN_;
    } else {          // aligned_b: qa[q] vs kb[p]/vb[p]; cosine vs va[q]
      Qg = qa + (size_t)q * N_ * IN_;
      Kg = kb + (size_t)p * N_ * IN_;
      Vg = vb + (size_t)p * N_ * IN_;
      Og = va + (size_t)q * N_ * IN_;
    }
    __syncthreads();
    // Q/K/Vo: contiguous 12.8 KB each -> async b128 copies
    for (int i = tid; i < (N_ * IN_) / 8; i += 256) {
      g2l_b128(Qg + i * 8, Qh + i * 8);
      g2l_b128(Kg + i * 8, Kh + i * 8);
      g2l_b128(Og + i * 8, Vo + i * 8);
    }
    for (int i = N_ * IN_ + tid; i < NPAD * IN_; i += 256) {
      Qh[i] = (_Float16)0.0f;
      Kh[i] = (_Float16)0.0f;
      Vo[i] = (_Float16)0.0f;
    }
    // V transposed into LDS: Vt[i][m] = V[m][i], zero-padded m >= 100
    for (int t = tid; t < KPAD * IN_; t += 256) {
      const int m = t >> 6, irow = t & 63;  // coalesced global reads
      Vt[irow * KPAD + m] = (m < N_) ? Vg[m * IN_ + irow] : (_Float16)0.0f;
    }
    wait_async();
    __syncthreads();

    // S = Q @ K^T (7x7 tiles, K=64 -> 2 WMMAs/tile)
    for (int t = wave; t < 49; t += 8) {
      const int tm = t / 7, tn = t % 7;
      v8f c = {};
#pragma unroll
      for (int k0 = 0; k0 < IN_; k0 += 32) {
        v16h a = frag_A(Qh + tm * 16 * IN_ + k0, IN_);
        v16h b = frag_B_T(Kh + tn * 16 * IN_ + k0, IN_);
        c = wmma32(a, b, c);
      }
      store_D_f32(S + tm * 16 * KPAD + tn * 16, KPAD, c);
    }
    __syncthreads();

    // Unnormalized softmax (cosine is scale-invariant): P = exp(scale*(S-max))
    for (int r = tid; r < NPAD; r += 256) {
      float m = -1e30f;
      for (int j = 0; j < N_; ++j) m = fmaxf(m, S[r * KPAD + j]);
      for (int j = 0; j < N_; ++j)
        P[r * KPAD + j] = (_Float16)__expf((S[r * KPAD + j] - m) * scale);
      for (int j = N_; j < KPAD; ++j) P[r * KPAD + j] = (_Float16)0.0f;
    }
    __syncthreads();

    // AL = P @ V (7x4 tiles, K=128 -> 4 WMMAs/tile); B from transposed Vt
    for (int t = wave; t < 28; t += 8) {
      const int tm = t >> 2, tn = t & 3;
      v8f c = {};
#pragma unroll
      for (int k0 = 0; k0 < KPAD; k0 += 32) {
        v16h a = frag_A(P + tm * 16 * KPAD + k0, KPAD);
        v16h b = frag_B_T(Vt + tn * 16 * KPAD + k0, KPAD);
        c = wmma32(a, b, c);
      }
      store_D_f32(AL + tm * 16 * IN_ + tn * 16, IN_, c);
    }
    __syncthreads();

    // cosine over rows 0..99
    float local = 0.0f;
    for (int r = tid; r < N_; r += 256) {
      float d = 0.0f, na = 0.0f, nv = 0.0f;
      for (int i = 0; i < IN_; ++i) {
        const float x = AL[r * IN_ + i];
        const float y = (float)Vo[r * IN_ + i];
        d += x * y;
        na += x * x;
        nv += y * y;
      }
      local += d / (fmaxf(sqrtf(na), 1e-8f) * fmaxf(sqrtf(nv), 1e-8f));
    }
    if (local != 0.0f) atomicAdd(&accRed, local);
    __syncthreads();
  }
  if (tid == 0) out[(size_t)p * B_ + q] = accRed * (1.0f / N_);
}

// ---------------------------------------------------------------------------
extern "C" void kernel_launch(void* const* d_in, const int* in_sizes, int n_in,
                              void* d_out, int out_size, void* d_ws,
                              size_t ws_size, hipStream_t stream) {
  const float* fa  = (const float*)d_in[0];
  const float* fb  = (const float*)d_in[1];
  const float* Wq1 = (const float*)d_in[2];
  const float* Wq2 = (const float*)d_in[3];
  const float* Wk1 = (const float*)d_in[4];
  const float* Wk2 = (const float*)d_in[5];
  const float* Wv1 = (const float*)d_in[6];
  const float* Wv2 = (const float*)d_in[7];
  float* out = (float*)d_out;

  _Float16* Xa   = (_Float16*)d_ws;                 // [6400,512]
  _Float16* Xb   = Xa + (size_t)M_ * C_;            // [6400,512]
  _Float16* out6 = Xb + (size_t)M_ * C_;            // 6 x [6400,64]
  _Float16* W1t  = out6 + (size_t)6 * M_ * IN_;     // 3 x [512,512] (n-major)
  _Float16* W2t  = W1t + (size_t)3 * C_ * C_;       // 3 x [64,512]  (n-major)

  const size_t nPack = (size_t)M_ * C_;
  pack_kernel<<<(unsigned)((nPack + 255) / 256), 256, 0, stream>>>(fa, fb, Xa,
                                                                   Xb);
  cvt_w_kernel<<<(3 * C_ * C_ + 255) / 256, 256, 0, stream>>>(
      Wq1, Wk1, Wv1, Wq2, Wk2, Wv2, W1t, W2t);

  proj_kernel<<<dim3(M_ / 16, 6), 256, 0, stream>>>(Xa, Xb, W1t, W2t, out6);

  const _Float16* qa = out6 + 0ull * M_ * IN_;
  const _Float16* ka = out6 + 1ull * M_ * IN_;
  const _Float16* va = out6 + 2ull * M_ * IN_;
  const _Float16* qb = out6 + 3ull * M_ * IN_;
  const _Float16* kb = out6 + 4ull * M_ * IN_;
  const _Float16* vb = out6 + 5ull * M_ * IN_;

  const size_t smem = (size_t)(2 * NPAD * IN_ + IN_ * KPAD + NPAD * IN_) *
                          sizeof(_Float16)                      // Qh,Kh,Vt,Vo
                    + (size_t)NPAD * KPAD * sizeof(float)        // S / AL
                    + (size_t)NPAD * KPAD * sizeof(_Float16);    // P
  attn_kernel<<<dim3(B_, B_), 256, smem, stream>>>(qa, ka, va, qb, kb, vb, out);
}
